// TripletCosineLoss_40956808135241
// MI455X (gfx1250) — matrix operand order repaired and checked by reading, not compiled, not run on hardware
//
#include <hip/hip_runtime.h>
#include <math.h>

#define MARGIN 0.3f
#define DDIM   128

typedef __attribute__((ext_vector_type(16))) _Float16     v16h;
typedef __attribute__((ext_vector_type(8)))  float        v8f;
typedef __attribute__((ext_vector_type(4)))  unsigned int v4u;
typedef __attribute__((ext_vector_type(4)))  _Float16     v4h;

union Frag16 {
    v16h h;
    v4u  u[2];
};

// ---------------------------------------------------------------------------
// Kernel 1: L2-normalize rows of [rows, 128] f32 -> f16.
// One wave per row: 4 floats per lane, shfl-xor sum-of-squares reduction.
// ---------------------------------------------------------------------------
__global__ __launch_bounds__(256)
void k_normalize_f32_to_f16(const float* __restrict__ in,
                            _Float16* __restrict__ out,
                            int rows) {
    const int gwave = (int)((blockIdx.x * blockDim.x + threadIdx.x) >> 5);
    const int lane  = (int)(threadIdx.x & 31);
    if (gwave >= rows) return;

    const float4 v = ((const float4*)(in + (size_t)gwave * DDIM))[lane];
    float ss = v.x * v.x + v.y * v.y + v.z * v.z + v.w * v.w;
#pragma unroll
    for (int off = 16; off > 0; off >>= 1)
        ss += __shfl_xor(ss, off, 32);

    const float inv = 1.0f / fmaxf(sqrtf(ss), 1e-12f);

    v4h h;
    h.x = (_Float16)(v.x * inv);
    h.y = (_Float16)(v.y * inv);
    h.z = (_Float16)(v.z * inv);
    h.w = (_Float16)(v.w * inv);
    *(v4h*)(out + (size_t)gwave * DDIM + lane * 4) = h;
}

// ---------------------------------------------------------------------------
// Kernel 2: fused sim-GEMM + masked min/max epilogue.
// Block = 8 waves, one 16-row anchor tile per block; wave w walks N-tiles
// w, w+8, ... A-fragments (4 k-steps) live in registers for the whole loop.
// ---------------------------------------------------------------------------
__global__ __launch_bounds__(256)
void k_fused_sim_minmax(const _Float16* __restrict__ a16,
                        const _Float16* __restrict__ s16,
                        const int* __restrict__ alab,
                        const int* __restrict__ slab,
                        float* __restrict__ rowloss,
                        int ntTotal) {
    __shared__ float posLds[8][16];
    __shared__ float negLds[8][16];

    const int lane     = (int)(threadIdx.x & 31);
    const int wave     = (int)(threadIdx.x >> 5);
    const int mtile    = (int)blockIdx.x;
    const int laneHalf = lane >> 4;   // 0: rows 0-7 / K-lo, 1: rows 8-15 / K-hi
    const int lmod     = lane & 15;

    // ---- A fragments: row = mtile*16 + (lane&15), kept in VGPRs ----
    const int arow = mtile * 16 + lmod;
    const v4u* abase = (const v4u*)(a16 + (size_t)arow * DDIM);
    Frag16 afr[4];
#pragma unroll
    for (int ks = 0; ks < 4; ++ks) {
        const int h0 = ks * 32 + laneHalf * 8;        // half-element offset
        afr[ks].u[0] = abase[(h0 >> 3)];              // VGPRs 0-3: K = h0..h0+7
        afr[ks].u[1] = abase[(h0 >> 3) + 2];          // VGPRs 4-7: K = h0+16..+23
    }

    // Anchor labels for the 8 rows this lane's C-fragment covers.
    int myAlab[8];
#pragma unroll
    for (int r = 0; r < 8; ++r)
        myAlab[r] = alab[mtile * 16 + laneHalf * 8 + r];

    float posmin[8], negmax[8];
#pragma unroll
    for (int r = 0; r < 8; ++r) { posmin[r] = INFINITY; negmax[r] = -INFINITY; }

    // ---- main loop over sample tiles ----
    for (int nt = wave; nt < ntTotal; nt += 8) {
        const int scol = nt * 16 + lmod;              // B column = sample row
        const v4u* bbase = (const v4u*)(s16 + (size_t)scol * DDIM);

        if (nt + 8 < ntTotal) {   // wave-uniform; prefetch next tile's column
            __builtin_prefetch(s16 + (size_t)(nt + 8) * 16 * DDIM + (size_t)lmod * DDIM, 0, 3);
        }

        v8f c = {};
#pragma unroll
        for (int ks = 0; ks < 4; ++ks) {
            Frag16 bfr;
            const int h0 = ks * 32 + laneHalf * 8;    // K = h0..h0+7, h0+16..h0+23
            bfr.u[0] = bbase[(h0 >> 3)];
            bfr.u[1] = bbase[(h0 >> 3) + 2];
            c = __builtin_amdgcn_wmma_f32_16x16x32_f16(
                    false, afr[ks].h, false, bfr.h, (short)0, c, false, false);
        }

        const int sl = slab[scol];
#pragma unroll
        for (int r = 0; r < 8; ++r) {
            const float v = c[r];
            if (myAlab[r] == sl) posmin[r] = fminf(posmin[r], v);
            else                 negmax[r] = fmaxf(negmax[r], v);
        }
    }

    // ---- reduce across the 16 lanes of each half-wave (columns) ----
#pragma unroll
    for (int r = 0; r < 8; ++r) {
#pragma unroll
        for (int off = 1; off < 16; off <<= 1) {
            posmin[r] = fminf(posmin[r], __shfl_xor(posmin[r], off, 32));
            negmax[r] = fmaxf(negmax[r], __shfl_xor(negmax[r], off, 32));
        }
    }

    if (lmod == 0) {
#pragma unroll
        for (int r = 0; r < 8; ++r) {
            posLds[wave][laneHalf * 8 + r] = posmin[r];
            negLds[wave][laneHalf * 8 + r] = negmax[r];
        }
    }
    __syncthreads();

    // ---- combine the 8 waves' partials, emit per-row loss ----
    if (threadIdx.x < 16) {
        float pm = INFINITY, nm = -INFINITY;
#pragma unroll
        for (int w = 0; w < 8; ++w) {
            pm = fminf(pm, posLds[w][threadIdx.x]);
            nm = fmaxf(nm, negLds[w][threadIdx.x]);
        }
        rowloss[mtile * 16 + threadIdx.x] = fmaxf(0.0f, nm - pm + MARGIN);
    }
}

// ---------------------------------------------------------------------------
// Kernel 3: deterministic mean over rowloss (single block, fixed-order tree).
// ---------------------------------------------------------------------------
__global__ __launch_bounds__(256)
void k_reduce_mean(const float* __restrict__ rowloss, float* __restrict__ out, int n) {
    __shared__ float sm[256];
    float s = 0.0f;
    for (int i = (int)threadIdx.x; i < n; i += 256) s += rowloss[i];
    sm[threadIdx.x] = s;
    __syncthreads();
#pragma unroll
    for (int off = 128; off > 0; off >>= 1) {
        if ((int)threadIdx.x < off) sm[threadIdx.x] += sm[threadIdx.x + off];
        __syncthreads();
    }
    if (threadIdx.x == 0) out[0] = sm[0] / (float)n;
}

// ---------------------------------------------------------------------------
extern "C" void kernel_launch(void* const* d_in, const int* in_sizes, int n_in,
                              void* d_out, int out_size, void* d_ws, size_t ws_size,
                              hipStream_t stream) {
    const float* avec = (const float*)d_in[0];
    const int*   alab = (const int*)d_in[1];
    const float* svec = (const float*)d_in[2];
    const int*   slab = (const int*)d_in[3];

    const int A = in_sizes[0] / DDIM;   // 8192
    const int S = in_sizes[2] / DDIM;   // 8192

    _Float16* a16     = (_Float16*)d_ws;
    _Float16* s16     = a16 + (size_t)A * DDIM;
    float*    rowloss = (float*)(s16 + (size_t)S * DDIM);

    // 8 rows per block (1 wave per row)
    k_normalize_f32_to_f16<<<(A + 7) / 8, 256, 0, stream>>>(avec, a16, A);
    k_normalize_f32_to_f16<<<(S + 7) / 8, 256, 0, stream>>>(svec, s16, S);

    // one block per 16-row anchor tile
    k_fused_sim_minmax<<<A / 16, 256, 0, stream>>>(a16, s16, alab, slab,
                                                   rowloss, S / 16);

    k_reduce_mean<<<1, 256, 0, stream>>>(rowloss, (float*)d_out, A);
}
